// PerformerModule_4690104287417
// MI455X (gfx1250) — matrix operand order repaired and checked by reading, not compiled, not run on hardware
//
#include <hip/hip_runtime.h>
#include <math.h>

// ---------------------------------------------------------------------------
// Performer forward for MI455X (gfx1250): register-blocked bf16 WMMA GEMMs
// (4x2 tiles / wave, 21 FLOP/B fragment intensity) + chunked FAVOR+ causal
// linear attention with WMMA intra-chunk scores.
// ---------------------------------------------------------------------------

typedef __bf16 bf16;
typedef bf16  v16bf __attribute__((ext_vector_type(16)));
typedef bf16  v8bf  __attribute__((ext_vector_type(8)));
typedef float v8f   __attribute__((ext_vector_type(8)));

constexpr int Bb  = 2;
constexpr int Ss  = 2048;
constexpr int Ee  = 512;
constexpr int Hh  = 8;
constexpr int Ll  = 4;
constexpr int DHd = 64;
constexpr int Mf  = 128;          // FAVOR features
constexpr int Tt  = Bb * Ss;      // 4096 tokens
constexpr int FF  = 4 * Ee;       // 2048
constexpr int NVo = 17;           // vocab+1

constexpr float DN    = 0.35355339059327373f;  // DH^-0.25
constexpr float RATIO = 0.08838834764831845f;  // M^-0.5
constexpr float KEPS  = 1e-4f;
constexpr float DEPS  = 1e-6f;

// register blocking: each wave computes (16*MT) x (16*NT) of C
constexpr int MT = 4;
constexpr int NT = 2;

// ---- WMMA fragment loaders (CDNA5 ISA 7.12.2, wave32) ---------------------
// A 16x32 bf16: lane group (lane>=16) selects K-halves {8..15,24..31};
// elements 0..7 = K(kh..kh+7), 8..15 = K(16+kh..16+kh+7).
__device__ __forceinline__ v16bf load_a_frag(const bf16* p) {
  v8bf lo = *(const v8bf*)(p);
  v8bf hi = *(const v8bf*)(p + 16);
  v16bf r;
#pragma unroll
  for (int i = 0; i < 8; ++i) { r[i] = lo[i]; r[i + 8] = hi[i]; }
  return r;
}
// B 32x16 bf16: same K order as A (lane>=16 selects K {8..15,24..31}).
__device__ __forceinline__ v16bf load_b_frag(const bf16* p) {
  v8bf lo = *(const v8bf*)(p);
  v8bf hi = *(const v8bf*)(p + 16);
  v16bf r;
#pragma unroll
  for (int i = 0; i < 8; ++i) { r[i] = lo[i]; r[i + 8] = hi[i]; }
  return r;
}

// ---- generic bf16 WMMA GEMM: C = act(A @ Bt^T + bias)*scale (+add) --------
// A:  Mdim x Kdim bf16 (row major, lda), optional per-z offset (head batching)
// Bt: Ndim x Kdim bf16 (i.e. B transposed, contiguous K)
// Mdim must be a multiple of 16*MT; Ndim is bounds-checked per 16-column tile.
template <bool GELU>
__global__ __launch_bounds__(256) void gemm_bf16(
    const bf16* __restrict__ A, int lda, long long azoff,
    const bf16* __restrict__ Bt,
    const float* __restrict__ bias,
    const float* __restrict__ add, int ldadd,
    float* __restrict__ C, int ldc, long long czoff,
    bf16* __restrict__ Cbf, int ldcb,
    int Mdim, int Ndim, int Kdim, float scale) {
  const int lane  = threadIdx.x & 31;
  const int wave  = threadIdx.x >> 5;
  const int tilesN = (Ndim + 16 * NT - 1) / (16 * NT);
  const int tilesM = Mdim / (16 * MT);
  const int t = blockIdx.x * 8 + wave;
  if (t >= tilesM * tilesN) return;             // wave-uniform: EXEC stays full
  const int tm = t / tilesN, tn = t % tilesN;
  const int z  = blockIdx.z;
  const int lrow = lane & 15;
  const int ah  = (lane >> 4) << 3;             // A K-half offset (8)
  const int bh2 = (lane >> 4) << 3;             // B K-half offset (8)

  const bf16* abase = A + (long long)z * azoff + ah;
  const bf16* ap[MT];
#pragma unroll
  for (int mt = 0; mt < MT; ++mt)
    ap[mt] = abase + (long long)((tm * MT + mt) * 16 + lrow) * lda;
  const bf16* bp[NT];
#pragma unroll
  for (int nt = 0; nt < NT; ++nt) {
    int c = (tn * NT + nt) * 16 + lrow;
    if (c >= Ndim) c = Ndim - 1;                // clamp: masked on store
    bp[nt] = Bt + (long long)c * Kdim + bh2;
  }

  v8f acc[MT][NT] = {};
  for (int kk = 0; kk < Kdim; kk += 32) {
    __builtin_prefetch(ap[0] + kk + 128, 0, 1);
    __builtin_prefetch(bp[0] + kk + 128, 0, 1);
    v16bf af[MT];
    v16bf bfr[NT];
#pragma unroll
    for (int mt = 0; mt < MT; ++mt) af[mt] = load_a_frag(ap[mt] + kk);
#pragma unroll
    for (int nt = 0; nt < NT; ++nt) bfr[nt] = load_b_frag(bp[nt] + kk);
#pragma unroll
    for (int mt = 0; mt < MT; ++mt)
#pragma unroll
      for (int nt = 0; nt < NT; ++nt)
        acc[mt][nt] = __builtin_amdgcn_wmma_f32_16x16x32_bf16(
            false, af[mt], false, bfr[nt], (short)0, acc[mt][nt], false, false);
  }

  const int mhi = (lane >> 4) << 3;             // D: lanes>=16 hold M=r+8
#pragma unroll
  for (int nt = 0; nt < NT; ++nt) {
    const int col = (tn * NT + nt) * 16 + lrow;
    if (col >= Ndim) continue;
    const float bvv = bias ? bias[col] : 0.f;
#pragma unroll
    for (int mt = 0; mt < MT; ++mt) {
#pragma unroll
      for (int r = 0; r < 8; ++r) {
        const int row = (tm * MT + mt) * 16 + r + mhi;
        float v = acc[mt][nt][r] + bvv;
        if (GELU) v = 0.5f * v * (1.f + erff(v * 0.70710678118f));
        v *= scale;
        if (add) v += add[(long long)row * ldadd + col];
        if (C)   C[(long long)z * czoff + (long long)row * ldc + col] = v;
        if (Cbf) Cbf[(long long)row * ldcb + col] = (bf16)v;
      }
    }
  }
}

// ---- embedding + right-shift with SOS -------------------------------------
__global__ void embed_kernel(const int* __restrict__ value,
                             const int* __restrict__ depth,
                             const int* __restrict__ pos,
                             const float* __restrict__ sos,
                             const float* __restrict__ tok,
                             const float* __restrict__ dep,
                             const float* __restrict__ spa,
                             float* __restrict__ x) {
  const int t = blockIdx.x;               // token = b*S + s
  const int s = t & (Ss - 1);
  float* xr = x + (long long)t * Ee;
  if (s == 0) {
#pragma unroll
    for (int e = threadIdx.x; e < Ee; e += 256) xr[e] = sos[e];
    return;
  }
  const int p = t - 1;                    // same b, previous s
  const int v  = value[p], dpt = depth[p];
  const int p0 = pos[p * 3 + 0], p1 = pos[p * 3 + 1], p2 = pos[p * 3 + 2];
#pragma unroll
  for (int e = threadIdx.x; e < Ee; e += 256) {
    xr[e] = tok[(long long)v * Ee + e] + dep[(long long)dpt * Ee + e]
          + spa[((long long)0 * 65 + p0) * Ee + e]
          + spa[((long long)1 * 65 + p1) * Ee + e]
          + spa[((long long)2 * 65 + p2) * Ee + e];
  }
}

// ---- layernorm: f32 in, bf16 out ------------------------------------------
__global__ __launch_bounds__(256) void layernorm_kernel(
    const float* __restrict__ x, const float* __restrict__ w,
    const float* __restrict__ b, bf16* __restrict__ out) {
  __shared__ float red[256];
  const int row = blockIdx.x, tid = threadIdx.x;
  const float* xr = x + (long long)row * Ee;
  const float v0 = xr[tid], v1 = xr[tid + 256];
  red[tid] = v0 + v1;
  __syncthreads();
  for (int o = 128; o > 0; o >>= 1) {
    if (tid < o) red[tid] += red[tid + o];
    __syncthreads();
  }
  const float mu = red[0] * (1.f / Ee);
  __syncthreads();
  const float d0 = v0 - mu, d1 = v1 - mu;
  red[tid] = d0 * d0 + d1 * d1;
  __syncthreads();
  for (int o = 128; o > 0; o >>= 1) {
    if (tid < o) red[tid] += red[tid + o];
    __syncthreads();
  }
  const float rs = rsqrtf(red[0] * (1.f / Ee) + 1e-5f);
  out[(long long)row * Ee + tid]       = (bf16)(d0 * rs * w[tid] + b[tid]);
  out[(long long)row * Ee + tid + 256] = (bf16)(d1 * rs * w[tid + 256] + b[tid + 256]);
}

// ---- weight convert / transpose to bf16 N x K -----------------------------
__global__ void transpose_to_bf16(const float* __restrict__ src,
                                  bf16* __restrict__ dst, int K, int N) {
  const int n = blockIdx.x * 16 + threadIdx.x;
  const int k = blockIdx.y * 16 + threadIdx.y;
  if (n < N && k < K) dst[(long long)n * K + k] = (bf16)src[(long long)k * N + n];
}

__global__ void conv_bf16(const float* __restrict__ s, bf16* __restrict__ d, int n) {
  const int i = blockIdx.x * 256 + threadIdx.x;
  if (i < n) d[i] = (bf16)s[i];
}

// ---- diag = 0.5 * sum(xd^2) per (token, head); input already scaled by dn --
__global__ __launch_bounds__(64) void diag_kernel(const bf16* __restrict__ q,
                                                  float* __restrict__ diag) {
  __shared__ float red[64];
  const int r = blockIdx.x;               // token*H + h
  const int token = r >> 3, h = r & 7;
  const float v = (float)q[(long long)token * Ee + h * DHd + threadIdx.x];
  red[threadIdx.x] = v * v;
  __syncthreads();
  for (int o = 32; o > 0; o >>= 1) {
    if (threadIdx.x < o) red[threadIdx.x] += red[threadIdx.x + o];
    __syncthreads();
  }
  if (threadIdx.x == 0) diag[r] = 0.5f * red[0];
}

// ---- FAVOR+ feature finalize: phi = ratio*(exp(dd - diag - mx) + eps) -----
__global__ __launch_bounds__(128) void feat_finalize(
    const float* __restrict__ dd, const float* __restrict__ diag,
    const float* __restrict__ gmax, bf16* __restrict__ out, int isQuery) {
  __shared__ float red[128];
  const int r = blockIdx.x, tid = threadIdx.x;
  const float v = dd[(long long)r * Mf + tid];
  float mx;
  if (isQuery) {                           // per-row max (query path)
    red[tid] = v;
    __syncthreads();
    for (int o = 64; o > 0; o >>= 1) {
      if (tid < o) red[tid] = fmaxf(red[tid], red[tid + o]);
      __syncthreads();
    }
    mx = red[0];
  } else {
    mx = gmax[0];                          // global scalar max (key path)
  }
  out[(long long)r * Mf + tid] = (bf16)(RATIO * (expf(v - diag[r] - mx) + KEPS));
}

// ---- deterministic two-stage global max reduction -------------------------
__global__ __launch_bounds__(256) void rmax_stage1(const float* __restrict__ d,
                                                   float* __restrict__ part, int n) {
  __shared__ float red[256];
  float m = -3.4e38f;
  for (long long i = (long long)blockIdx.x * 256 + threadIdx.x; i < n;
       i += (long long)gridDim.x * 256)
    m = fmaxf(m, d[i]);
  red[threadIdx.x] = m;
  __syncthreads();
  for (int o = 128; o > 0; o >>= 1) {
    if (threadIdx.x < o) red[threadIdx.x] = fmaxf(red[threadIdx.x], red[threadIdx.x + o]);
    __syncthreads();
  }
  if (threadIdx.x == 0) part[blockIdx.x] = red[0];
}

__global__ __launch_bounds__(256) void rmax_stage2(const float* __restrict__ part,
                                                   float* __restrict__ outv, int n) {
  __shared__ float red[256];
  float m = -3.4e38f;
  for (int i = threadIdx.x; i < n; i += 256) m = fmaxf(m, part[i]);
  red[threadIdx.x] = m;
  __syncthreads();
  for (int o = 128; o > 0; o >>= 1) {
    if (threadIdx.x < o) red[threadIdx.x] = fmaxf(red[threadIdx.x], red[threadIdx.x + o]);
    __syncthreads();
  }
  if (threadIdx.x == 0) outv[0] = red[0];
}

// ---- chunked causal linear attention: one wave per (b,h) ------------------
// State S[128x64] f32 + z[128] in LDS; chunk=16 tokens; P = Qc Kc^T via WMMA.
__global__ void attn_kernel(const bf16* __restrict__ qp,
                            const bf16* __restrict__ kp,
                            const float* __restrict__ v,
                            bf16* __restrict__ obf) {
  __shared__ float Sst[Mf * DHd];   // 32 KB
  __shared__ float zst[Mf];
  __shared__ float qs[16 * Mf];
  __shared__ float ks[16 * Mf];
  __shared__ float vs[16 * DHd];
  __shared__ float ps[16 * 16];
  __shared__ float den[16];

  const int lane = threadIdx.x;
  const int bh = blockIdx.x;
  const int b = bh >> 3, h = bh & 7;
  for (int i = lane; i < Mf * DHd; i += 32) Sst[i] = 0.f;
  for (int i = lane; i < Mf; i += 32) zst[i] = 0.f;
  __syncthreads();

  const int lrow = lane & 15;
  const int ah = (lane >> 4) << 3;
  const int bhalf = (lane >> 4) << 3;
  const int mhi = (lane >> 4) << 3;

  for (int c = 0; c < Ss / 16; ++c) {
    const long long base = (long long)b * Ss + c * 16;
    // stage chunk into LDS (f32 working copies)
    for (int i = lane; i < 16 * Mf; i += 32) {
      const int rr = i >> 7, mm = i & (Mf - 1);
      qs[i] = (float)qp[(base + rr) * (Hh * Mf) + h * Mf + mm];
      ks[i] = (float)kp[(base + rr) * (Hh * Mf) + h * Mf + mm];
    }
    for (int i = lane; i < 16 * DHd; i += 32) {
      const int rr = i >> 6, dd = i & (DHd - 1);
      vs[i] = v[(base + rr) * Ee + h * DHd + dd];
    }
    // P = Qc [16x128] @ Kc^T [128x16] via 4 WMMA steps over K=features
    v8f p = {};
    const bf16* qrow = qp + (base + lrow) * (Hh * Mf) + h * Mf + ah;
    const bf16* krow = kp + (base + lrow) * (Hh * Mf) + h * Mf + bhalf;
#pragma unroll
    for (int kk = 0; kk < Mf; kk += 32) {
      v16bf a = load_a_frag(qrow + kk);
      v16bf bmat = load_b_frag(krow + kk);
      p = __builtin_amdgcn_wmma_f32_16x16x32_bf16(false, a, false, bmat,
                                                  (short)0, p, false, false);
    }
    // causal mask (include diagonal: scan adds k_t before producing out_t)
#pragma unroll
    for (int r = 0; r < 8; ++r) {
      const int i = r + mhi, j = lrow;
      ps[i * 16 + j] = (j <= i) ? p[r] : 0.f;
    }
    __syncthreads();
    // denominators: q_i . z_prev + sum_{j<=i} P_ij + eps
    if (lane < 16) {
      float s = DEPS;
      for (int m = 0; m < Mf; ++m) s += qs[lane * Mf + m] * zst[m];
      for (int j = 0; j <= lane; ++j) s += ps[lane * 16 + j];
      den[lane] = s;
    }
    __syncthreads();
    // out_i = (q_i . S_prev + sum_{j<=i} P_ij v_j) / den_i
    for (int i = 0; i < 16; ++i) {
      const float inv = 1.f / den[i];
#pragma unroll
      for (int half = 0; half < 2; ++half) {
        const int d = lane + half * 32;
        float num = 0.f;
        for (int m = 0; m < Mf; ++m) num += qs[i * Mf + m] * Sst[m * DHd + d];
        for (int j = 0; j <= i; ++j) num += ps[i * 16 + j] * vs[j * DHd + d];
        obf[(base + i) * Ee + h * DHd + d] = (bf16)(num * inv);
      }
    }
    __syncthreads();
    // rank-16 state update
    for (int idx = lane; idx < Mf * DHd; idx += 32) {
      const int m = idx >> 6, d = idx & (DHd - 1);
      float a = Sst[idx];
#pragma unroll
      for (int j = 0; j < 16; ++j) a += ks[j * Mf + m] * vs[j * DHd + d];
      Sst[idx] = a;
    }
    for (int m = lane; m < Mf; m += 32) {
      float a = zst[m];
#pragma unroll
      for (int j = 0; j < 16; ++j) a += ks[j * Mf + m];
      zst[m] = a;
    }
    __syncthreads();
  }
}

// ---------------------------------------------------------------------------
static void gemm_launch(hipStream_t st, const bf16* A, int lda, long long azoff,
                        const bf16* Bt, const float* bias,
                        const float* add, int ldadd,
                        float* C, int ldc, long long czoff,
                        bf16* Cbf, int ldcb,
                        int M, int N, int K, float scale, bool gelu, int Z) {
  const int tilesN = (N + 16 * NT - 1) / (16 * NT);
  const int tilesM = M / (16 * MT);
  const int tiles = tilesM * tilesN;
  dim3 grid((tiles + 7) / 8, 1, Z), blk(256);
  if (gelu)
    gemm_bf16<true><<<grid, blk, 0, st>>>(A, lda, azoff, Bt, bias, add, ldadd,
                                          C, ldc, czoff, Cbf, ldcb, M, N, K, scale);
  else
    gemm_bf16<false><<<grid, blk, 0, st>>>(A, lda, azoff, Bt, bias, add, ldadd,
                                           C, ldc, czoff, Cbf, ldcb, M, N, K, scale);
}

extern "C" void kernel_launch(void* const* d_in, const int* in_sizes, int n_in,
                              void* d_out, int out_size, void* d_ws, size_t ws_size,
                              hipStream_t stream) {
  (void)in_sizes; (void)n_in; (void)out_size; (void)ws_size;
  const int*   value = (const int*)d_in[0];
  const int*   depth = (const int*)d_in[1];
  const int*   pos   = (const int*)d_in[2];
  const float* proj  = (const float*)d_in[3];
  const float* sos   = (const float*)d_in[4];
  const float* tok   = (const float*)d_in[5];
  const float* dep   = (const float*)d_in[6];
  const float* spa   = (const float*)d_in[7];
  const float* ln1w  = (const float*)d_in[8];
  const float* ln1b  = (const float*)d_in[9];
  const float* Wq    = (const float*)d_in[10];
  const float* bq    = (const float*)d_in[11];
  const float* Wk    = (const float*)d_in[12];
  const float* bk    = (const float*)d_in[13];
  const float* Wv    = (const float*)d_in[14];
  const float* bv    = (const float*)d_in[15];
  const float* Wo    = (const float*)d_in[16];
  const float* bo    = (const float*)d_in[17];
  const float* ln2w  = (const float*)d_in[18];
  const float* ln2b  = (const float*)d_in[19];
  const float* W1    = (const float*)d_in[20];
  const float* b1    = (const float*)d_in[21];
  const float* W2    = (const float*)d_in[22];
  const float* b2    = (const float*)d_in[23];
  const float* headw = (const float*)d_in[24];
  float* out = (float*)d_out;

  // ---- workspace carve-up (fixed, deterministic) ----
  char* wsb = (char*)d_ws;
  size_t off = 0;
  auto take = [&](size_t bytes) -> char* {
    char* p = wsb + off;
    off = (off + bytes + 255) & ~(size_t)255;
    return p;
  };
  bf16* WqT   = (bf16*)take((size_t)Ll * Ee * Ee * 2);
  bf16* WkT   = (bf16*)take((size_t)Ll * Ee * Ee * 2);
  bf16* WvT   = (bf16*)take((size_t)Ll * Ee * Ee * 2);
  bf16* WoT   = (bf16*)take((size_t)Ll * Ee * Ee * 2);
  bf16* W1T   = (bf16*)take((size_t)Ll * FF * Ee * 2);   // [2048][512] per layer
  bf16* W2T   = (bf16*)take((size_t)Ll * Ee * FF * 2);   // [512][2048] per layer
  bf16* projT = (bf16*)take((size_t)Ll * Mf * DHd * 2);  // already N x K
  bf16* headT = (bf16*)take((size_t)32 * Ee * 2);        // padded rows
  float* x    = (float*)take((size_t)Tt * Ee * 4);
  bf16* hbf   = (bf16*)take((size_t)Tt * Ee * 2);
  bf16* qbf   = (bf16*)take((size_t)Tt * Ee * 2);        // dn-scaled
  bf16* kbf   = (bf16*)take((size_t)Tt * Ee * 2);        // dn-scaled
  float* vbuf = (float*)take((size_t)Tt * Ee * 4);
  float* ddb  = (float*)take((size_t)Tt * Hh * Mf * 4);  // shared q/k
  float* diag = (float*)take((size_t)Tt * Hh * 4);
  bf16* qp    = (bf16*)take((size_t)Tt * Hh * Mf * 2);
  bf16* kp    = (bf16*)take((size_t)Tt * Hh * Mf * 2);
  bf16* obf   = (bf16*)take((size_t)Tt * Ee * 2);
  bf16* h1bf  = (bf16*)take((size_t)Tt * FF * 2);
  float* part = (float*)take((size_t)2048 * 4);
  float* kmax = (float*)take(4);

  // ---- weight conversion / transposition to bf16 N x K ----
  for (int l = 0; l < Ll; ++l) {
    dim3 g(Ee / 16, Ee / 16), b(16, 16);
    transpose_to_bf16<<<g, b, 0, stream>>>(Wq + (size_t)l * Ee * Ee, WqT + (size_t)l * Ee * Ee, Ee, Ee);
    transpose_to_bf16<<<g, b, 0, stream>>>(Wk + (size_t)l * Ee * Ee, WkT + (size_t)l * Ee * Ee, Ee, Ee);
    transpose_to_bf16<<<g, b, 0, stream>>>(Wv + (size_t)l * Ee * Ee, WvT + (size_t)l * Ee * Ee, Ee, Ee);
    transpose_to_bf16<<<g, b, 0, stream>>>(Wo + (size_t)l * Ee * Ee, WoT + (size_t)l * Ee * Ee, Ee, Ee);
    transpose_to_bf16<<<dim3(FF / 16, Ee / 16), b, 0, stream>>>(
        W1 + (size_t)l * Ee * FF, W1T + (size_t)l * FF * Ee, Ee, FF);
    transpose_to_bf16<<<dim3(Ee / 16, FF / 16), b, 0, stream>>>(
        W2 + (size_t)l * FF * Ee, W2T + (size_t)l * Ee * FF, FF, Ee);
  }
  transpose_to_bf16<<<dim3(2, Ee / 16), dim3(16, 16), 0, stream>>>(headw, headT, Ee, NVo);
  {
    const int n = Ll * Mf * DHd;
    conv_bf16<<<(n + 255) / 256, 256, 0, stream>>>(proj, projT, n);
  }

  // ---- embeddings + SOS shift ----
  embed_kernel<<<Tt, 256, 0, stream>>>(value, depth, pos, sos, tok, dep, spa, x);

  const int nDD = Tt * Hh * Mf;
  for (int l = 0; l < Ll; ++l) {
    // LN1 -> bf16
    layernorm_kernel<<<Tt, 256, 0, stream>>>(x, ln1w + (size_t)l * Ee, ln1b + (size_t)l * Ee, hbf);
    // QKV projections (q,k scaled by dn for the feature map)
    gemm_launch(stream, hbf, Ee, 0, WqT + (size_t)l * Ee * Ee, bq + (size_t)l * Ee,
                nullptr, 0, nullptr, 0, 0, qbf, Ee, Tt, Ee, Ee, DN, false, 1);
    gemm_launch(stream, hbf, Ee, 0, WkT + (size_t)l * Ee * Ee, bk + (size_t)l * Ee,
                nullptr, 0, nullptr, 0, 0, kbf, Ee, Tt, Ee, Ee, DN, false, 1);
    gemm_launch(stream, hbf, Ee, 0, WvT + (size_t)l * Ee * Ee, bv + (size_t)l * Ee,
                nullptr, 0, vbuf, Ee, 0, nullptr, 0, Tt, Ee, Ee, 1.f, false, 1);
    // Q features: dd = xd @ proj^T per head (grid.z = heads), then row-max exp
    diag_kernel<<<Tt * Hh, 64, 0, stream>>>(qbf, diag);
    gemm_launch(stream, qbf, Ee, DHd, projT + (size_t)l * Mf * DHd, nullptr,
                nullptr, 0, ddb, Hh * Mf, Mf, nullptr, 0, Tt, Mf, DHd, 1.f, false, Hh);
    feat_finalize<<<Tt * Hh, 128, 0, stream>>>(ddb, diag, kmax, qp, 1);
    // K features: global scalar max
    diag_kernel<<<Tt * Hh, 64, 0, stream>>>(kbf, diag);
    gemm_launch(stream, kbf, Ee, DHd, projT + (size_t)l * Mf * DHd, nullptr,
                nullptr, 0, ddb, Hh * Mf, Mf, nullptr, 0, Tt, Mf, DHd, 1.f, false, Hh);
    rmax_stage1<<<2048, 256, 0, stream>>>(ddb, part, nDD);
    rmax_stage2<<<1, 256, 0, stream>>>(part, kmax, 2048);
    feat_finalize<<<Tt * Hh, 128, 0, stream>>>(ddb, diag, kmax, kp, 0);
    // causal linear attention scan (chunked, WMMA intra-chunk scores)
    attn_kernel<<<Bb * Hh, 32, 0, stream>>>(qp, kp, vbuf, obf);
    // out projection + residual
    gemm_launch(stream, obf, Ee, 0, WoT + (size_t)l * Ee * Ee, bo + (size_t)l * Ee,
                x, Ee, x, Ee, 0, nullptr, 0, Tt, Ee, Ee, 1.f, false, 1);
    // FFN
    layernorm_kernel<<<Tt, 256, 0, stream>>>(x, ln2w + (size_t)l * Ee, ln2b + (size_t)l * Ee, hbf);
    gemm_launch(stream, hbf, Ee, 0, W1T + (size_t)l * FF * Ee, b1 + (size_t)l * FF,
                nullptr, 0, nullptr, 0, 0, h1bf, FF, Tt, FF, Ee, 1.f, true, 1);
    gemm_launch(stream, h1bf, FF, 0, W2T + (size_t)l * Ee * FF, b2 + (size_t)l * Ee,
                x, Ee, x, Ee, 0, obf, Ee, Tt, Ee, FF, 1.f, false, 1);
    // obf now holds bf16(x) — consumed by next layer's attention output or head
  }
  // head: logits = x @ head_w  (N=17, bounds-checked tiles)
  gemm_launch(stream, obf, Ee, 0, headT, nullptr, nullptr, 0,
              out, NVo, 0, nullptr, 0, Tt, NVo, Ee, 1.f, false, 1);
}
